// Block_4440996184362
// MI455X (gfx1250) — matrix operand order repaired and checked by reading, not compiled, not run on hardware
//
#include <hip/hip_runtime.h>
#include <math.h>

typedef __attribute__((ext_vector_type(16))) __bf16 v16bf;
typedef __attribute__((ext_vector_type(8)))  float  v8f;

#define CDIM 1024
#define TDIM 2048
#define BDIM 2
#define HDIM 16
#define MDIM (BDIM * TDIM)   // 4096
#define HD   64

union U32x8 { unsigned u[8]; v16bf v; };

// A-matrix (16x32 bf16) per-lane K index for vgpr-slot v (0..7), half = lane/16
__device__ __forceinline__ int kpatA(int v, int hf) { return 16 * hf + 2 * v; }
// B-matrix (32x16 bf16) per-lane K index (contiguous k = 16*half + 2*v, same order as A)
__device__ __forceinline__ int kpatB(int v, int hf) { return 16 * hf + 2 * v; }

// ------------------------------------------------ fused transpose + fp32->bf16
// in: [K][N] fp32 row-major;  out: [N][K] bf16 row-major
__global__ __launch_bounds__(256)
void cvtT_bf16(const float* __restrict__ in, __bf16* __restrict__ out, int K, int N) {
    __shared__ float tile[16][17];
    const int tx = threadIdx.x & 15, ty = threadIdx.x >> 4;
    const int n0 = blockIdx.x * 16, k0 = blockIdx.y * 16;
    tile[ty][tx] = in[(size_t)(k0 + ty) * N + n0 + tx];
    __syncthreads();
    out[(size_t)(n0 + ty) * K + k0 + tx] = (__bf16)tile[tx][ty];
}

// ------------------------------------------------ layernorm -> bf16
__global__ __launch_bounds__(256)
void ln_bf16(const float* __restrict__ x, const float* __restrict__ g,
             const float* __restrict__ b, __bf16* __restrict__ out) {
    __shared__ float r1[256], r2[256];
    const int tid = threadIdx.x;
    const size_t row = blockIdx.x;
    const float* xr = x + row * CDIM;
    float vals[4];
    float s = 0.f, s2 = 0.f;
#pragma unroll
    for (int i = 0; i < 4; i++) {
        float vv = xr[tid + i * 256];
        vals[i] = vv; s += vv; s2 += vv * vv;
    }
    r1[tid] = s; r2[tid] = s2;
    __syncthreads();
    for (int o = 128; o > 0; o >>= 1) {
        if (tid < o) { r1[tid] += r1[tid + o]; r2[tid] += r2[tid + o]; }
        __syncthreads();
    }
    float mu  = r1[0] * (1.0f / CDIM);
    float var = r2[0] * (1.0f / CDIM) - mu * mu;
    float rs  = rsqrtf(var + 1e-5f);
#pragma unroll
    for (int i = 0; i < 4; i++) {
        int idx = tid + i * 256;
        out[row * CDIM + idx] = (__bf16)((vals[i] - mu) * rs * g[idx] + b[idx]);
    }
}

// ------------------------------------------------ bf16 GEMM (WMMA + async LDS copy)
// A: [Mn][Kn] bf16;  Wt: [Nn][Kn] bf16 (pre-transposed weight)
// Out[m,n] = epi( sum_k A[m,k]*Wt[n,k] + bias[n] )
// flags bit0: exact GELU
// flags bit1: bf16 head-scatter out [B,H,T,hd]
// flags bit2: bf16 head-scatter out [B,H,hd,T] (for V)
__global__ __launch_bounds__(256)
void gemm_bf16(const __bf16* __restrict__ A, const __bf16* __restrict__ Wt,
               const float* __restrict__ bias, const float* __restrict__ resid,
               float* __restrict__ outf, __bf16* __restrict__ outb,
               int Mn, int Nn, int Kn, int flags, float scale) {
    __shared__ unsigned As[128 * 16];   // 128 rows x 32 k (bf16 pairs)
    __shared__ unsigned Bs[128 * 16];   // 128 cols x 32 k (bf16 pairs)

    const int tid  = threadIdx.x;
    const int lane = tid & 31, wid = tid >> 5;
    const int lr = lane & 15, hf = lane >> 4;
    const int wm = wid & 3, wn = wid >> 2;           // 4 x 2 wave grid
    const int m0 = blockIdx.y * 128;
    const int n0 = blockIdx.x * 128;

    v8f acc[2][4] = {};
    const char* Aby = (const char*)A;
    const char* Bby = (const char*)Wt;
    const unsigned ldsA = (unsigned)(size_t)(&As[0]);
    const unsigned ldsB = (unsigned)(size_t)(&Bs[0]);
    const size_t rstride = (size_t)Kn * 2;           // bytes per row

    for (int kb = 0; kb < Kn; kb += 32) {
        if (kb + 32 < Kn)
            __builtin_prefetch(Bby + (size_t)(n0 + (tid >> 1)) * rstride + (size_t)(kb + 32) * 2, 0, 1);

        // async DMA: 8KB A tile + 8KB B tile, 16B chunks, VGPR-bypassing
#pragma unroll
        for (int i = 0; i < 2; i++) {
            int c   = tid + i * 256;       // chunk 0..511
            int row = c >> 2;              // 4 x 16B chunks per 64B row
            int cb  = (c & 3) * 16;
            unsigned long long ga = (unsigned long long)(Aby + (size_t)(m0 + row) * rstride + (size_t)kb * 2 + cb);
            unsigned long long gb = (unsigned long long)(Bby + (size_t)(n0 + row) * rstride + (size_t)kb * 2 + cb);
            unsigned la = ldsA + (unsigned)(c * 16);
            unsigned lb = ldsB + (unsigned)(c * 16);
            asm volatile("global_load_async_to_lds_b128 %0, %1, off" :: "v"(la), "v"(ga) : "memory");
            asm volatile("global_load_async_to_lds_b128 %0, %1, off" :: "v"(lb), "v"(gb) : "memory");
        }
        asm volatile("s_wait_asynccnt 0" ::: "memory");
        __syncthreads();

        U32x8 af[2], bf4[4];
#pragma unroll
        for (int mt = 0; mt < 2; mt++) {
            int row = wm * 32 + mt * 16 + lr;
#pragma unroll
            for (int v = 0; v < 8; v++) af[mt].u[v] = As[row * 16 + (kpatA(v, hf) >> 1)];
        }
#pragma unroll
        for (int nt = 0; nt < 4; nt++) {
            int n = wn * 64 + nt * 16 + lr;
#pragma unroll
            for (int v = 0; v < 8; v++) bf4[nt].u[v] = Bs[n * 16 + (kpatB(v, hf) >> 1)];
        }
#pragma unroll
        for (int mt = 0; mt < 2; mt++)
#pragma unroll
            for (int nt = 0; nt < 4; nt++)
                acc[mt][nt] = __builtin_amdgcn_wmma_f32_16x16x32_bf16(
                    false, af[mt].v, false, bf4[nt].v, (short)0, acc[mt][nt], false, false);
        __syncthreads();
    }

    // epilogue
#pragma unroll
    for (int mt = 0; mt < 2; mt++) {
#pragma unroll
        for (int nt = 0; nt < 4; nt++) {
            int cg = n0 + wn * 64 + nt * 16 + lr;
            float bvv = bias ? bias[cg] : 0.f;
#pragma unroll
            for (int r = 0; r < 8; r++) {
                int rg = m0 + wm * 32 + mt * 16 + r + 8 * hf;
                float vv = acc[mt][nt][r] + bvv;
                if (flags & 1) vv = 0.5f * vv * (1.0f + erff(vv * 0.70710678118654752f));
                if (resid) vv += resid[(size_t)rg * Nn + cg];
                if (outf) outf[(size_t)rg * Nn + cg] = vv;
                if (outb) {
                    float ov = vv * scale;
                    int bq = rg / TDIM, tq = rg % TDIM;
                    int hh = cg >> 6, dd = cg & 63;
                    if (flags & 2)
                        outb[((((size_t)bq * HDIM + hh) * TDIM + tq) << 6) + dd] = (__bf16)ov;
                    else if (flags & 4)
                        outb[(((size_t)bq * HDIM + hh) * HD + dd) * TDIM + tq] = (__bf16)ov;
                    else
                        outb[(size_t)rg * Nn + cg] = (__bf16)ov;
                }
            }
        }
    }
}

// ------------------------------------------------ flash attention (WMMA)
// q,k: [B,H,T,64] bf16 (q pre-scaled by 1/8);  v: [B,H,64,T] bf16;  yb: [B,T,C] bf16
__global__ __launch_bounds__(128)
void attn_kernel(const __bf16* __restrict__ q, const __bf16* __restrict__ k,
                 const __bf16* __restrict__ v, __bf16* __restrict__ yb) {
    __shared__ __bf16 Pl[4][16][32];   // per-wave P staging (16 rows x 32 keys)
    const int lane = threadIdx.x & 31, wid = threadIdx.x >> 5;
    const int lr = lane & 15, hf = lane >> 4;
    const int bh = blockIdx.x;
    const int bb = bh >> 4;            // H == 16
    const int hh = bh & 15;
    const int qt = blockIdx.y * 4 + wid;
    const int q0 = qt * 16;

    const unsigned* qg  = (const unsigned*)(q + ((size_t)bh * TDIM + q0) * HD);
    const unsigned* kg  = (const unsigned*)(k + (size_t)bh * TDIM * HD);
    const unsigned* vgT = (const unsigned*)(v + (size_t)bh * HD * TDIM);   // [64][T]

    // q fragments: two K=32 steps over hd=64
    U32x8 aq[2];
#pragma unroll
    for (int s = 0; s < 2; s++)
#pragma unroll
        for (int vv = 0; vv < 8; vv++) {
            int d = s * 32 + kpatA(vv, hf);
            aq[s].u[vv] = qg[lr * 32 + (d >> 1)];
        }

    v8f acc[4] = {};
    float mrow[8], lrow[8];
#pragma unroll
    for (int r = 0; r < 8; r++) { mrow[r] = -3.0e38f; lrow[r] = 0.f; }

    const int nch = (q0 + 16 + 31) >> 5;   // causal: keys < q0+16
    for (int c = 0; c < nch; c++) {
        const int kb = c * 32;
        v8f s01[2];
#pragma unroll
        for (int t2 = 0; t2 < 2; t2++) {
            int key = kb + t2 * 16 + lr;
            U32x8 bk0, bk1;
#pragma unroll
            for (int vv = 0; vv < 8; vv++) {
                bk0.u[vv] = kg[key * 32 + ((     kpatB(vv, hf)) >> 1)];
                bk1.u[vv] = kg[key * 32 + ((32 + kpatB(vv, hf)) >> 1)];
            }
            v8f sc = {};
            sc = __builtin_amdgcn_wmma_f32_16x16x32_bf16(false, aq[0].v, false, bk0.v, (short)0, sc, false, false);
            sc = __builtin_amdgcn_wmma_f32_16x16x32_bf16(false, aq[1].v, false, bk1.v, (short)0, sc, false, false);
#pragma unroll
            for (int r = 0; r < 8; r++) {
                int rg = q0 + r + 8 * hf;
                if (key > rg) sc[r] = -3.0e38f;
            }
            s01[t2] = sc;
        }

        // online softmax on C-layout rows, cross-lane over 16-lane halves
        float mx[8], ps[8], alpha[8];
#pragma unroll
        for (int r = 0; r < 8; r++) mx[r] = fmaxf(s01[0][r], s01[1][r]);
#pragma unroll
        for (int off = 1; off <= 8; off <<= 1)
#pragma unroll
            for (int r = 0; r < 8; r++) mx[r] = fmaxf(mx[r], __shfl_xor(mx[r], off, 32));
#pragma unroll
        for (int r = 0; r < 8; r++) {
            float mn = fmaxf(mrow[r], mx[r]);
            alpha[r] = expf(mrow[r] - mn);
            float p0 = expf(s01[0][r] - mn);
            float p1 = expf(s01[1][r] - mn);
            Pl[wid][r + 8 * hf][lr]      = (__bf16)p0;
            Pl[wid][r + 8 * hf][16 + lr] = (__bf16)p1;
            ps[r] = p0 + p1;
            mrow[r] = mn;
        }
#pragma unroll
        for (int off = 1; off <= 8; off <<= 1)
#pragma unroll
            for (int r = 0; r < 8; r++) ps[r] += __shfl_xor(ps[r], off, 32);
#pragma unroll
        for (int r = 0; r < 8; r++) lrow[r] = lrow[r] * alpha[r] + ps[r];
#pragma unroll
        for (int nt = 0; nt < 4; nt++)
#pragma unroll
            for (int r = 0; r < 8; r++) acc[nt][r] *= alpha[r];

        asm volatile("s_wait_dscnt 0" ::: "memory");

        // reload P in A-fragment layout
        U32x8 ap;
        const unsigned* Pr = (const unsigned*)&Pl[wid][lr][0];
#pragma unroll
        for (int vv = 0; vv < 8; vv++) ap.u[vv] = Pr[kpatA(vv, hf) >> 1];

        // y += P @ V   (V transposed: contiguous u32 along keys)
#pragma unroll
        for (int nt = 0; nt < 4; nt++) {
            int dd = nt * 16 + lr;
            U32x8 bvf;
#pragma unroll
            for (int vv = 0; vv < 8; vv++)
                bvf.u[vv] = vgT[(size_t)dd * (TDIM >> 1) + (kb >> 1) + 8 * hf + vv];
            acc[nt] = __builtin_amdgcn_wmma_f32_16x16x32_bf16(
                false, ap.v, false, bvf.v, (short)0, acc[nt], false, false);
        }
        asm volatile("s_wait_dscnt 0" ::: "memory");
    }

    float inv[8];
#pragma unroll
    for (int r = 0; r < 8; r++) inv[r] = 1.0f / lrow[r];
#pragma unroll
    for (int nt = 0; nt < 4; nt++)
#pragma unroll
        for (int r = 0; r < 8; r++) {
            int tg  = q0 + r + 8 * hf;
            int col = hh * 64 + nt * 16 + lr;
            yb[((size_t)bb * TDIM + tg) * CDIM + col] = (__bf16)(acc[nt][r] * inv[r]);
        }
}

// ------------------------------------------------ launch
extern "C" void kernel_launch(void* const* d_in, const int* in_sizes, int n_in,
                              void* d_out, int out_size, void* d_ws, size_t ws_size,
                              hipStream_t stream) {
    const float* x     = (const float*)d_in[0];
    const float* ln1_g = (const float*)d_in[1];
    const float* ln1_b = (const float*)d_in[2];
    const float* Wq    = (const float*)d_in[3];
    const float* bq    = (const float*)d_in[4];
    const float* Wk    = (const float*)d_in[5];
    const float* bk    = (const float*)d_in[6];
    const float* Wv    = (const float*)d_in[7];
    const float* bv    = (const float*)d_in[8];
    const float* Wp    = (const float*)d_in[9];
    const float* bp    = (const float*)d_in[10];
    const float* ln2_g = (const float*)d_in[11];
    const float* ln2_b = (const float*)d_in[12];
    const float* W1    = (const float*)d_in[13];
    const float* b1    = (const float*)d_in[14];
    const float* W2    = (const float*)d_in[15];
    const float* b2    = (const float*)d_in[16];
    float* out = (float*)d_out;

    char* ws = (char*)d_ws;
    size_t off = 0;
    auto alloc = [&](size_t bytes) -> void* {
        void* p = ws + off;
        off += (bytes + 255) & ~(size_t)255;
        return p;
    };
    __bf16* hb   = (__bf16*)alloc((size_t)MDIM * CDIM * 2);
    __bf16* qbuf = (__bf16*)alloc((size_t)MDIM * CDIM * 2);
    __bf16* kbuf = (__bf16*)alloc((size_t)MDIM * CDIM * 2);
    __bf16* vbuf = (__bf16*)alloc((size_t)MDIM * CDIM * 2);
    __bf16* ybuf = (__bf16*)alloc((size_t)MDIM * CDIM * 2);
    float*  x1   = (float*)alloc((size_t)MDIM * CDIM * 4);
    __bf16* h2b  = (__bf16*)alloc((size_t)MDIM * CDIM * 2);
    __bf16* mb   = (__bf16*)alloc((size_t)MDIM * 4 * CDIM * 2);
    __bf16* wqt  = (__bf16*)alloc((size_t)CDIM * CDIM * 2);
    __bf16* wkt  = (__bf16*)alloc((size_t)CDIM * CDIM * 2);
    __bf16* wvt  = (__bf16*)alloc((size_t)CDIM * CDIM * 2);
    __bf16* wpt  = (__bf16*)alloc((size_t)CDIM * CDIM * 2);
    __bf16* w1t  = (__bf16*)alloc((size_t)CDIM * 4 * CDIM * 2);
    __bf16* w2t  = (__bf16*)alloc((size_t)CDIM * 4 * CDIM * 2);

    // transpose+convert weights: Wt[n][k] = (bf16)W[k][n]
    dim3 tb(256);
    cvtT_bf16<<<dim3(CDIM / 16, CDIM / 16), tb, 0, stream>>>(Wq, wqt, CDIM, CDIM);
    cvtT_bf16<<<dim3(CDIM / 16, CDIM / 16), tb, 0, stream>>>(Wk, wkt, CDIM, CDIM);
    cvtT_bf16<<<dim3(CDIM / 16, CDIM / 16), tb, 0, stream>>>(Wv, wvt, CDIM, CDIM);
    cvtT_bf16<<<dim3(CDIM / 16, CDIM / 16), tb, 0, stream>>>(Wp, wpt, CDIM, CDIM);
    cvtT_bf16<<<dim3(4 * CDIM / 16, CDIM / 16), tb, 0, stream>>>(W1, w1t, CDIM, 4 * CDIM);
    cvtT_bf16<<<dim3(CDIM / 16, 4 * CDIM / 16), tb, 0, stream>>>(W2, w2t, 4 * CDIM, CDIM);

    ln_bf16<<<MDIM, 256, 0, stream>>>(x, ln1_g, ln1_b, hb);

    dim3 g8(CDIM / 128, MDIM / 128);
    // QKV head-scatter; q scaled by 1/sqrt(64); v stored [B,H,hd,T]
    gemm_bf16<<<g8, 256, 0, stream>>>(hb, wqt, bq, nullptr, nullptr, qbuf,
                                      MDIM, CDIM, CDIM, 2, 0.125f);
    gemm_bf16<<<g8, 256, 0, stream>>>(hb, wkt, bk, nullptr, nullptr, kbuf,
                                      MDIM, CDIM, CDIM, 2, 1.0f);
    gemm_bf16<<<g8, 256, 0, stream>>>(hb, wvt, bv, nullptr, nullptr, vbuf,
                                      MDIM, CDIM, CDIM, 4, 1.0f);

    attn_kernel<<<dim3(BDIM * HDIM, TDIM / 64), 128, 0, stream>>>(qbuf, kbuf, vbuf, ybuf);

    // proj + residual(x) -> x1 (fp32)
    gemm_bf16<<<g8, 256, 0, stream>>>(ybuf, wpt, bp, x, x1, nullptr,
                                      MDIM, CDIM, CDIM, 0, 1.0f);

    ln_bf16<<<MDIM, 256, 0, stream>>>(x1, ln2_g, ln2_b, h2b);

    dim3 g32(4 * CDIM / 128, MDIM / 128);
    // MLP up + exact GELU -> mb (bf16)
    gemm_bf16<<<g32, 256, 0, stream>>>(h2b, w1t, b1, nullptr, nullptr, mb,
                                       MDIM, 4 * CDIM, CDIM, 1, 1.0f);
    // MLP down + residual(x1) -> out (fp32)
    gemm_bf16<<<g8, 256, 0, stream>>>(mb, w2t, b2, x1, out, nullptr,
                                      MDIM, CDIM, 4 * CDIM, 0, 1.0f);
}
